// FunctionNPairLoss_51891794870946
// MI455X (gfx1250) — matrix operand order repaired and checked by reading, not compiled, not run on hardware
//
#include <hip/hip_runtime.h>
#include <hip/hip_bf16.h>

// ---------------------------------------------------------------------------
// N-pair loss, fused WMMA GEMM + online log-sum-exp (gfx1250 / CDNA5, wave32)
// ---------------------------------------------------------------------------

#define NN 8192
#define DD 512
#define NCLS 128
#define CSPL 8                    // column chunks (grid.y)
#define CCHUNK (NN / CSPL)        // 1024 columns per block
#define CAP 512                   // per-class member capacity (mean 64, 56 sigma margin)

typedef __attribute__((ext_vector_type(8)))  float  v8f;
typedef __attribute__((ext_vector_type(16))) __bf16 v16bf;

// -------------------- helpers --------------------

__device__ __forceinline__ unsigned short f2bf_rne(float x) {
  unsigned int u = __float_as_uint(x);
  unsigned int r = (u + 0x7FFFu + ((u >> 16) & 1u)) >> 16;
  return (unsigned short)r;
}
__device__ __forceinline__ float bf2f(unsigned short h) {
  return __uint_as_float(((unsigned int)h) << 16);
}

// A fragment, 16x32 bf16 (M x K): lane m=l&15; VGPR0-3: K = k0 + (l>=16?8:0)+0..7
//                                  VGPR4-7: K = k0+16+(l>=16?8:0)+0..7
__device__ __forceinline__ v16bf load_frag_a(const unsigned short* row, int k0, int o8) {
  union { v16bf v; uint4 q[2]; } u;
  u.q[0] = *(const uint4*)(row + k0 + o8);
  u.q[1] = *(const uint4*)(row + k0 + 16 + o8);
  return u.v;
}
// B fragment, 32x16 bf16 (K x N): lane n=l&15; VGPR0-3: K = k0 + (l>=16?8:0)+0..7, VGPR4-7: K = k0+16+(l>=16?8:0)+0..7
__device__ __forceinline__ v16bf load_frag_b(const unsigned short* row, int k0, int o16) {
  union { v16bf v; uint4 q[2]; } u;
  u.q[0] = *(const uint4*)(row + k0 + o16);
  u.q[1] = *(const uint4*)(row + k0 + o16 + 16);
  return u.v;
}

#define WMMA_BF16(A, B, C) \
  __builtin_amdgcn_wmma_f32_16x16x32_bf16(false, (A), false, (B), (short)0, (C), false, false)

// -------------------- kernel 1: fp32 -> (hi,lo) bf16 split + row norms -----

__global__ __launch_bounds__(256) void prep_kernel(const float* __restrict__ E,
                                                   unsigned short* __restrict__ Ehi,
                                                   unsigned short* __restrict__ Elo,
                                                   float* __restrict__ sq) {
  const int lane = threadIdx.x & 31;
  const int wave = threadIdx.x >> 5;
  const int row  = blockIdx.x * 8 + wave;
  const float* er = E + (size_t)row * DD;
  float acc = 0.f;
  for (int k = lane; k < DD; k += 32) {
    float x = er[k];
    unsigned short h = f2bf_rne(x);
    float xl = x - bf2f(h);
    Ehi[(size_t)row * DD + k] = h;
    Elo[(size_t)row * DD + k] = f2bf_rne(xl);
    acc = fmaf(x, x, acc);
  }
#pragma unroll
  for (int m = 16; m; m >>= 1) acc += __shfl_xor(acc, m, 32);
  if (lane == 0) sq[row] = acc;
}

// -------------------- kernel 2: fused GEMM + per-row online LSE ------------
// block = 128 threads (4 waves). Wave handles 32 rows; block handles 128 rows
// x CCHUNK columns. Split-bf16 product: g = hi*hi + hi*lo + lo*hi (fp32 acc).

__global__ __launch_bounds__(128) void npair_lse_kernel(
    const unsigned short* __restrict__ Ehi, const unsigned short* __restrict__ Elo,
    const float* __restrict__ sq, const int* __restrict__ lab,
    float2* __restrict__ partial) {
  const int lane   = threadIdx.x & 31;
  const int wave   = threadIdx.x >> 5;
  const int rowBlk = blockIdx.x * 128 + wave * 32;
  const int colBlk = blockIdx.y * CCHUNK;
  const int l15    = lane & 15;
  const int hi8    = (lane >> 4) * 8;
  const int o8     = hi8;            // A K-offset per lane half
  const int o16    = hi8 * 2;        // unused: B loads also take o8

  // 16 row-slots per lane: si = mt*8 + r  ->  row = rowBlk + mt*16 + r + hi8
  float m[16], s[16], rsq[16];
  int   rlab[16];
#pragma unroll
  for (int mt = 0; mt < 2; ++mt)
#pragma unroll
    for (int r = 0; r < 8; ++r) {
      const int si  = mt * 8 + r;
      const int row = rowBlk + mt * 16 + r + hi8;
      m[si] = -3.4e38f;  // strictly below mask value -3.0e38
      s[si] = 0.f;
      rsq[si]  = sq[row];
      rlab[si] = lab[row];
    }

  const unsigned short* a0h = Ehi + (size_t)(rowBlk + l15) * DD;
  const unsigned short* a1h = Ehi + (size_t)(rowBlk + 16 + l15) * DD;
  const unsigned short* a0l = Elo + (size_t)(rowBlk + l15) * DD;
  const unsigned short* a1l = Elo + (size_t)(rowBlk + 16 + l15) * DD;

  for (int ct = 0; ct < CCHUNK / 64; ++ct) {
    const int cb = colBlk + ct * 64;
    const unsigned short* brh[4];
    const unsigned short* brl[4];
#pragma unroll
    for (int nt = 0; nt < 4; ++nt) {
      const int c = cb + nt * 16 + l15;
      brh[nt] = Ehi + (size_t)c * DD;
      brl[nt] = Elo + (size_t)c * DD;
    }
    v8f acc[2][4];
#pragma unroll
    for (int mt = 0; mt < 2; ++mt)
#pragma unroll
      for (int nt = 0; nt < 4; ++nt) acc[mt][nt] = (v8f)(0.f);

    for (int kk = 0; kk < DD; kk += 32) {
      const v16bf fa0h = load_frag_a(a0h, kk, o8);
      const v16bf fa1h = load_frag_a(a1h, kk, o8);
      const v16bf fa0l = load_frag_a(a0l, kk, o8);
      const v16bf fa1l = load_frag_a(a1l, kk, o8);
#pragma unroll
      for (int nt = 0; nt < 4; ++nt) {
        const v16bf fbh = load_frag_b(brh[nt], kk, o8);
        const v16bf fbl = load_frag_b(brl[nt], kk, o8);
        acc[0][nt] = WMMA_BF16(fa0h, fbh, acc[0][nt]);
        acc[0][nt] = WMMA_BF16(fa0h, fbl, acc[0][nt]);
        acc[0][nt] = WMMA_BF16(fa0l, fbh, acc[0][nt]);
        acc[1][nt] = WMMA_BF16(fa1h, fbh, acc[1][nt]);
        acc[1][nt] = WMMA_BF16(fa1h, fbl, acc[1][nt]);
        acc[1][nt] = WMMA_BF16(fa1l, fbh, acc[1][nt]);
      }
    }

    // epilogue: dist = sq_i + sq_j - 2g; mask same-class; online LSE update
#pragma unroll
    for (int nt = 0; nt < 4; ++nt) {
      const int c  = cb + nt * 16 + l15;
      const float cs = sq[c];
      const int   cl = lab[c];
#pragma unroll
      for (int mt = 0; mt < 2; ++mt)
#pragma unroll
        for (int r = 0; r < 8; ++r) {
          const int si = mt * 8 + r;
          float d = fmaf(-2.f, acc[mt][nt][r], rsq[si] + cs);
          d = (rlab[si] == cl) ? -3.0e38f : d;
          if (d > m[si]) {             // rare after warm-up
            s[si] = s[si] * __expf(m[si] - d) + 1.f;
            m[si] = d;
          } else {
            s[si] += __expf(d - m[si]);
          }
        }
    }
  }

  // combine across the 16 lanes of each half (columns of the same row)
#pragma unroll
  for (int si = 0; si < 16; ++si) {
    float mm = m[si], ss = s[si];
#pragma unroll
    for (int msk = 1; msk < 16; msk <<= 1) {
      const float om = __shfl_xor(mm, msk, 32);
      const float os = __shfl_xor(ss, msk, 32);
      const float mn = fmaxf(mm, om);
      ss = ss * __expf(mm - mn) + os * __expf(om - mn);
      mm = mn;
    }
    if (l15 == 0) {
      const int row = rowBlk + (si >> 3) * 16 + (si & 7) + hi8;
      partial[(size_t)row * CSPL + blockIdx.y] = make_float2(mm, ss);
    }
  }
}

// -------------------- kernel 3: combine column chunks -> log_negsum --------

__global__ __launch_bounds__(128) void reduce_lns_kernel(const float2* __restrict__ partial,
                                                         float* __restrict__ lns) {
  const int row = blockIdx.x * blockDim.x + threadIdx.x;
  if (row >= NN) return;
  float m = -3.4e38f, s = 0.f;
  for (int c = 0; c < CSPL; ++c) {
    const float2 p = partial[(size_t)row * CSPL + c];
    const float mn = fmaxf(m, p.x);
    s = s * __expf(m - mn) + p.y * __expf(p.x - mn);
    m = mn;
  }
  lns[row] = m + __logf(s);
}

// -------------------- kernel 4: deterministic per-class member lists -------

__global__ __launch_bounds__(128) void build_members_kernel(const int* __restrict__ lab,
                                                            int* __restrict__ members,
                                                            int* __restrict__ counts) {
  const int c = threadIdx.x;  // 128 threads, one class each, sequential scan
  int cnt = 0;
  for (int i = 0; i < NN; ++i)
    if (lab[i] == c) {
      if (cnt < CAP) members[c * CAP + cnt] = i;
      ++cnt;
    }
  counts[c] = cnt < CAP ? cnt : CAP;
}

// -------------------- kernel 5: intra-class pair loss (fp32, tiny) ---------

__global__ __launch_bounds__(256) void pair_loss_kernel(
    const float* __restrict__ E, const float* __restrict__ sq,
    const float* __restrict__ lns, const int* __restrict__ members,
    const int* __restrict__ counts, float* __restrict__ class_sum,
    float* __restrict__ class_cnt) {
  const int c    = blockIdx.x;
  const int lane = threadIdx.x & 31;
  const int wave = threadIdx.x >> 5;
  const int cnt  = counts[c];
  __shared__ float wsum[8];
  float acc = 0.f;
  for (int a = wave; a < cnt; a += 8) {       // anchor (smaller index)
    const int ga = members[c * CAP + a];
    const float* ea = E + (size_t)ga * DD;
    const float za  = lns[ga];
    const float sqa = sq[ga];
    for (int b = a + 1; b < cnt; ++b) {
      const int gb = members[c * CAP + b];
      const float* eb = E + (size_t)gb * DD;
      float dot = 0.f;
      for (int k = lane; k < DD; k += 32) dot = fmaf(ea[k], eb[k], dot);
#pragma unroll
      for (int msk = 16; msk; msk >>= 1) dot += __shfl_xor(dot, msk, 32);
      const float dij = sqa + sq[gb] - 2.f * dot;
      const float z   = za - dij;
      const float pl  = fmaxf(z, 0.f) + __logf(1.f + __expf(-fabsf(z)));
      if (lane == 0) acc += pl;
    }
  }
  if (lane == 0) wsum[wave] = acc;
  __syncthreads();
  if (threadIdx.x == 0) {
    float t = 0.f;
    for (int w = 0; w < 8; ++w) t += wsum[w];
    class_sum[c] = t;
    class_cnt[c] = 0.5f * (float)cnt * (float)(cnt - 1);
  }
}

// -------------------- kernel 6: final scalar reduction ---------------------

__global__ __launch_bounds__(128) void finalize_kernel(const float* __restrict__ class_sum,
                                                       const float* __restrict__ class_cnt,
                                                       float* __restrict__ out, int out_size) {
  __shared__ float ssum[4];
  __shared__ int   scnt[4];
  const int t = threadIdx.x;
  const int lane = t & 31, wave = t >> 5;
  const float cc = class_cnt[t];
  float mean = (cc > 0.f) ? class_sum[t] / cc : 0.f;
  int   v    = (cc > 0.f) ? 1 : 0;
#pragma unroll
  for (int msk = 16; msk; msk >>= 1) {
    mean += __shfl_xor(mean, msk, 32);
    v    += __shfl_xor(v, msk, 32);
  }
  if (lane == 0) { ssum[wave] = mean; scnt[wave] = v; }
  __syncthreads();
  if (t == 0) {
    const float S = ssum[0] + ssum[1] + ssum[2] + ssum[3];
    const int   V = scnt[0] + scnt[1] + scnt[2] + scnt[3];
    out[0] = S / (float)V;
    if (out_size > 1) out[1] = (float)V;
  }
}

// -------------------- launcher ---------------------------------------------

extern "C" void kernel_launch(void* const* d_in, const int* in_sizes, int n_in,
                              void* d_out, int out_size, void* d_ws, size_t ws_size,
                              hipStream_t stream) {
  (void)in_sizes; (void)n_in; (void)ws_size;
  const float* E   = (const float*)d_in[0];
  const int*   lab = (const int*)d_in[1];
  float*       out = (float*)d_out;

  char* ws = (char*)d_ws;
  size_t off = 0;
  auto alloc = [&](size_t bytes) {
    void* p = ws + off;
    off = (off + bytes + 255) & ~(size_t)255;
    return p;
  };
  unsigned short* Ehi  = (unsigned short*)alloc((size_t)NN * DD * 2);
  unsigned short* Elo  = (unsigned short*)alloc((size_t)NN * DD * 2);
  float*   sq       = (float*)alloc((size_t)NN * 4);
  float*   lns      = (float*)alloc((size_t)NN * 4);
  float2*  partial  = (float2*)alloc((size_t)NN * CSPL * sizeof(float2));
  int*     members  = (int*)alloc((size_t)NCLS * CAP * 4);
  int*     counts   = (int*)alloc((size_t)NCLS * 4);
  float*   csum     = (float*)alloc((size_t)NCLS * 4);
  float*   ccnt     = (float*)alloc((size_t)NCLS * 4);

  prep_kernel<<<NN / 8, 256, 0, stream>>>(E, Ehi, Elo, sq);
  npair_lse_kernel<<<dim3(NN / 128, CSPL), 128, 0, stream>>>(Ehi, Elo, sq, lab, partial);
  reduce_lns_kernel<<<NN / 128, 128, 0, stream>>>(partial, lns);
  build_members_kernel<<<1, 128, 0, stream>>>(lab, members, counts);
  pair_loss_kernel<<<NCLS, 256, 0, stream>>>(E, sq, lns, members, counts, csum, ccnt);
  finalize_kernel<<<1, 128, 0, stream>>>(csum, ccnt, out, out_size);
}
